// PC_RNN_HC_M_14413910245619
// MI455X (gfx1250) — compile-verified, matrix operand order checked by reading, not run on hardware
//
#include <hip/hip_runtime.h>
#include <math.h>

#define T_LEN   1024
#define BATCH   512
#define CAUSES  32
#define STATES  256
#define OUTD    64
#define FACTOR  256
#define MT      16                 // batch rows per workgroup (WMMA M)
#define NWG     (BATCH / MT)       // 32 persistent workgroups
#define NTHREADS 256               // 8 wave32 waves

typedef __bf16 bf16;
typedef __bf16 bf16x8 __attribute__((ext_vector_type(8)));
typedef __bf16 v16bf  __attribute__((ext_vector_type(16)));
typedef float  v8f    __attribute__((ext_vector_type(8)));

// ---- bf16 weight workspace layout (element offsets) ----
#define WS_WPDT 0                                   // [F][S]  (w_pd transposed)
#define WS_WFD  (WS_WPDT + FACTOR * STATES)         // [S][F]  (w_fd as-is)
#define WS_WFDT (WS_WFD  + STATES * FACTOR)         // [F][S]  (w_fd transposed)
#define WS_WO   (WS_WFDT + FACTOR * STATES)         // [O][S]  (w_o as-is)
#define WS_WOT  (WS_WO   + OUTD   * STATES)         // [S][O]  (w_o transposed)
#define WS_WCD  (WS_WOT  + STATES * OUTD)           // [C][F]  (w_cd as-is)
#define WS_WCDT (WS_WCD  + CAUSES * FACTOR)         // [F][C]  (w_cd transposed)
#define WS_TOTAL (WS_WCDT + FACTOR * CAUSES)

// Dynamic LDS layout (bytes): weights 229376 + activations 63744 = 293120
#define SMEM_BYTES 293120

// ---------------------------------------------------------------------------
// Pre-pass: convert f32 weights -> bf16 (+ transposed copies) in workspace.
// ---------------------------------------------------------------------------
__global__ void convert_weights(const float* __restrict__ wo,
                                const float* __restrict__ wpd,
                                const float* __restrict__ wfd,
                                const float* __restrict__ wcd,
                                bf16* __restrict__ ws) {
    int i = blockIdx.x * blockDim.x + threadIdx.x;
    int n = gridDim.x * blockDim.x;
    for (int e = i; e < STATES * FACTOR; e += n) {
        int s = e / FACTOR, f = e % FACTOR;
        ws[WS_WPDT + f * STATES + s] = (bf16)wpd[e];
        ws[WS_WFD + e]               = (bf16)wfd[e];
        ws[WS_WFDT + f * STATES + s] = (bf16)wfd[e];
    }
    for (int e = i; e < OUTD * STATES; e += n) {
        int o = e / STATES, s = e % STATES;
        ws[WS_WO + e]             = (bf16)wo[e];
        ws[WS_WOT + s * OUTD + o] = (bf16)wo[e];
    }
    for (int e = i; e < CAUSES * FACTOR; e += n) {
        int c = e / FACTOR, f = e % FACTOR;
        ws[WS_WCD + e]               = (bf16)wcd[e];
        ws[WS_WCDT + f * CAUSES + c] = (bf16)wcd[e];
    }
}

// ---------------------------------------------------------------------------
// WMMA helpers. 16-bit operand layout (ISA 7.12.2):
//   lane<16  : row/col (lane&15), elems [0..7]=K 0..7,  [8..15]=K 16..23
//   lane>=16 : row/col (lane&15), elems [0..7]=K 8..15, [8..15]=K 24..31
// With a [N][K] (transposed) operand this is two contiguous 16-byte loads.
// ---------------------------------------------------------------------------
__device__ __forceinline__ v16bf load_tile(const bf16* p, int ldk, int row0,
                                           int k0, int lane) {
    int r  = lane & 15;
    int hi = (lane >> 4) & 1;
    const bf16* base = p + (row0 + r) * ldk + k0 + hi * 8;
    bf16x8 c0 = *(const bf16x8*)(base);
    bf16x8 c1 = *(const bf16x8*)(base + 16);
    v16bf out;
#pragma unroll
    for (int i = 0; i < 8; ++i) { out[i] = c0[i]; out[8 + i] = c1[i]; }
    return out;
}

__device__ __forceinline__ v8f gemm_acc(const bf16* A, int lda,
                                        const bf16* B, int ldb, int n0,
                                        int K, int lane, v8f acc) {
#pragma unroll
    for (int k0 = 0; k0 < K; k0 += 32) {
        v16bf a = load_tile(A, lda, 0, k0, lane);
        v16bf b = load_tile(B, ldb, n0, k0, lane);
        acc = __builtin_amdgcn_wmma_f32_16x16x32_bf16(false, a, false, b,
                                                      (short)0, acc, false, false);
    }
    return acc;
}

// D layout: lane<16 -> rows 0..7, lane>=16 -> rows 8..15; col = lane&15.
__device__ __forceinline__ void store_tile_f32(float* dst, int ld, int n0,
                                               int lane, v8f acc) {
    int col   = n0 + (lane & 15);
    int rbase = ((lane >> 4) & 1) * 8;
#pragma unroll
    for (int r = 0; r < 8; ++r) dst[(rbase + r) * ld + col] = acc[r];
}

__device__ __forceinline__ void copy_lds16(void* dst, const void* src,
                                           int bytes, int tid) {
    uint4* d = (uint4*)dst;
    const uint4* s = (const uint4*)src;
    int n = bytes >> 4;
    for (int i = tid; i < n; i += NTHREADS) d[i] = s[i];
}

// ---------------------------------------------------------------------------
// Persistent kernel: one WG = 16 batch rows, full 1024-step recurrence.
// ---------------------------------------------------------------------------
__global__ void __launch_bounds__(NTHREADS)
pc_rnn_kernel(const float* __restrict__ x, const float* __restrict__ c_init,
              const float* __restrict__ h_init, const float* __restrict__ b_o,
              const float* __restrict__ b_r, const bf16* __restrict__ ws,
              float* __restrict__ out) {
    extern __shared__ __align__(16) char smem[];
    // LDS-cached weights (224 KB)
    bf16* wpdT = (bf16*)smem;                 // [F][S]
    bf16* wo   = wpdT + FACTOR * STATES;      // [O][S]
    bf16* woT  = wo   + OUTD * STATES;        // [S][O]
    bf16* wcd  = woT  + STATES * OUTD;        // [C][F]
    bf16* wcdT = wcd  + CAUSES * FACTOR;      // [F][C]
    // Activations / state
    float* sH  = (float*)(wcdT + FACTOR * CAUSES); // [MT][S] h (prior/post)
    float* sP  = sH + MT * STATES;                 // [MT][F] pd
    float* sQ  = sP + MT * STATES;                 // [MT][F] cd / herr / ehf
    float* sC  = sQ + MT * STATES;                 // [MT][C] causes
    float* sBr = sC + MT * CAUSES;                 // [S]
    float* sBo = sBr + STATES;                     // [O]
    bf16*  sA  = (bf16*)(sBo + OUTD);              // [MT][S] bf16 A staging
    bf16*  sCb = sA + MT * STATES;                 // [MT][C] bf16 c
    bf16*  sEb = sCb + MT * CAUSES;                // [MT][O] bf16 error

    const int tid  = threadIdx.x;
    const int lane = tid & 31;
    const int wave = tid >> 5;
    const int b0   = blockIdx.x * MT;

    // one-time setup: weights into LDS, state from global
    copy_lds16(wpdT, ws + WS_WPDT, FACTOR * STATES * 2, tid);
    copy_lds16(wo,   ws + WS_WO,   OUTD * STATES * 2, tid);
    copy_lds16(woT,  ws + WS_WOT,  STATES * OUTD * 2, tid);
    copy_lds16(wcd,  ws + WS_WCD,  CAUSES * FACTOR * 2, tid);
    copy_lds16(wcdT, ws + WS_WCDT, FACTOR * CAUSES * 2, tid);
    for (int i = tid; i < MT * STATES; i += NTHREADS)
        sH[i] = h_init[(b0 + i / STATES) * STATES + (i % STATES)];
    for (int i = tid; i < MT * CAUSES; i += NTHREADS)
        sC[i] = c_init[(b0 + i / CAUSES) * CAUSES + (i % CAUSES)];
    for (int i = tid; i < STATES; i += NTHREADS) sBr[i] = b_r[i];
    for (int i = tid; i < OUTD; i += NTHREADS)  sBo[i] = b_o[i];
    __syncthreads();

    const bf16* wfd_g  = ws + WS_WFD;   // [S][F], streamed from L2
    const bf16* wfdT_g = ws + WS_WFDT;  // [F][S], streamed from L2

    for (int t = 0; t < T_LEN; ++t) {
        // P1: th_old = tanh(h) -> sA ; c -> bf16
        for (int i = tid; i < MT * STATES; i += NTHREADS)
            sA[i] = (bf16)tanhf(sH[i]);
        for (int i = tid; i < MT * CAUSES; i += NTHREADS)
            sCb[i] = (bf16)sC[i];
        __syncthreads();
        // P2: pd = th_old @ w_pd -> sP ; cd = c @ w_cd -> sQ
#pragma unroll
        for (int j = 0; j < 2; ++j) {
            int n0 = (2 * wave + j) * 16;
            v8f acc = {};
            acc = gemm_acc(sA, STATES, wpdT, STATES, n0, STATES, lane, acc);
            store_tile_f32(sP, FACTOR, n0, lane, acc);
            v8f acc2 = {};
            acc2 = gemm_acc(sCb, CAUSES, wcdT, CAUSES, n0, CAUSES, lane, acc2);
            store_tile_f32(sQ, FACTOR, n0, lane, acc2);
        }
        __syncthreads();
        // P3: A = bf16(pd * cd)
        for (int i = tid; i < MT * FACTOR; i += NTHREADS)
            sA[i] = (bf16)(sP[i] * sQ[i]);
        __syncthreads();
        // P4: drive = (pd*cd) @ w_fd^T ; h_prior = 0.9 h + 0.1 (drive + b_r)
#pragma unroll
        for (int j = 0; j < 2; ++j) {
            int n0 = (2 * wave + j) * 16;
            v8f acc = {};
            acc = gemm_acc(sA, FACTOR, wfd_g, FACTOR, n0, FACTOR, lane, acc);
            int col = n0 + (lane & 15);
            int rbase = ((lane >> 4) & 1) * 8;
#pragma unroll
            for (int r = 0; r < 8; ++r) {
                int idx = (rbase + r) * STATES + col;
                sH[idx] = 0.9f * sH[idx] + 0.1f * (acc[r] + sBr[col]);
            }
        }
        __syncthreads();
        // P5: th_prior -> sA
        for (int i = tid; i < MT * STATES; i += NTHREADS)
            sA[i] = (bf16)tanhf(sH[i]);
        __syncthreads();
        // P6: x_pred = th_prior @ w_o^T + b_o ; error -> out + sEb (waves 0-3)
        if (wave < 4) {
            int n0 = wave * 16;
            v8f acc = {};
            acc = gemm_acc(sA, STATES, wo, STATES, n0, STATES, lane, acc);
            int col = n0 + (lane & 15);
            int rbase = ((lane >> 4) & 1) * 8;
#pragma unroll
            for (int r = 0; r < 8; ++r) {
                int row = rbase + r;
                size_t g = ((size_t)t * BATCH + (b0 + row)) * OUTD + col;
                float e = acc[r] + sBo[col] - x[g];
                out[g] = e;
                sEb[row * OUTD + col] = (bf16)e;
            }
        }
        __syncthreads();
        // P7: herr = error @ w_o -> sQ
#pragma unroll
        for (int j = 0; j < 2; ++j) {
            int n0 = (2 * wave + j) * 16;
            v8f acc = {};
            acc = gemm_acc(sEb, OUTD, woT, OUTD, n0, OUTD, lane, acc);
            store_tile_f32(sQ, STATES, n0, lane, acc);
        }
        __syncthreads();
        // P8: corr = 0.1 (1 - thp^2) herr ; h_post = h_prior - corr ; error_h = corr
        for (int i = tid; i < MT * STATES; i += NTHREADS) {
            float thp = (float)sA[i];
            float corr = 0.1f * (1.0f - thp * thp) * sQ[i];
            sH[i] -= corr;
            sA[i] = (bf16)corr;
        }
        __syncthreads();
        // P9: ehf = error_h @ w_fd -> sQ
#pragma unroll
        for (int j = 0; j < 2; ++j) {
            int n0 = (2 * wave + j) * 16;
            v8f acc = {};
            acc = gemm_acc(sA, STATES, wfdT_g, STATES, n0, STATES, lane, acc);
            store_tile_f32(sQ, FACTOR, n0, lane, acc);
        }
        __syncthreads();
        // P10: A = bf16(pd * ehf)
        for (int i = tid; i < MT * FACTOR; i += NTHREADS)
            sA[i] = (bf16)(sP[i] * sQ[i]);
        __syncthreads();
        // P11: c -= 0.1 * ((pd*ehf) @ w_cd^T)   (waves 0-1, N=32)
        if (wave < 2) {
            int n0 = wave * 16;
            v8f acc = {};
            acc = gemm_acc(sA, FACTOR, wcd, FACTOR, n0, FACTOR, lane, acc);
            int col = n0 + (lane & 15);
            int rbase = ((lane >> 4) & 1) * 8;
#pragma unroll
            for (int r = 0; r < 8; ++r)
                sC[(rbase + r) * CAUSES + col] -= 0.1f * acc[r];
        }
        // prefetch next timestep's x tile into caches
        if (t + 1 < T_LEN && tid < MT)
            __builtin_prefetch(&x[((size_t)(t + 1) * BATCH + (b0 + tid)) * OUTD], 0, 3);
        __syncthreads();
    }
}

extern "C" void kernel_launch(void* const* d_in, const int* in_sizes, int n_in,
                              void* d_out, int out_size, void* d_ws, size_t ws_size,
                              hipStream_t stream) {
    (void)in_sizes; (void)n_in; (void)out_size; (void)ws_size;
    const float* x      = (const float*)d_in[0];
    const float* c_init = (const float*)d_in[1];
    const float* h_init = (const float*)d_in[2];
    const float* w_o    = (const float*)d_in[3];
    const float* b_o    = (const float*)d_in[4];
    const float* w_pd   = (const float*)d_in[5];
    const float* w_fd   = (const float*)d_in[6];
    const float* w_cd   = (const float*)d_in[7];
    const float* b_r    = (const float*)d_in[8];
    bf16*  ws  = (bf16*)d_ws;
    float* out = (float*)d_out;

    convert_weights<<<128, 256, 0, stream>>>(w_o, w_pd, w_fd, w_cd, ws);

    // allow >64KB dynamic LDS (ISA: up to 320KB per workgroup on CDNA5)
    hipFuncSetAttribute(reinterpret_cast<const void*>(pc_rnn_kernel),
                        hipFuncAttributeMaxDynamicSharedMemorySize, SMEM_BYTES);
    pc_rnn_kernel<<<NWG, NTHREADS, SMEM_BYTES, stream>>>(
        x, c_init, h_init, b_o, b_r, ws, out);
}
